// _Attention_65738769432879
// MI455X (gfx1250) — compile-verified, matrix operand order checked by reading, not run on hardware
//
#include <hip/hip_runtime.h>
#include <hip/hip_bf16.h>

typedef __bf16 bf16;
typedef __attribute__((ext_vector_type(8)))  bf16  v8bf;
typedef __attribute__((ext_vector_type(16))) bf16  v16bf;
typedef __attribute__((ext_vector_type(8)))  float v8f;

#define DIM   1024
#define NSEQ  2048
#define NB    4
#define HEADS 16
#define HD    64
#define SCALE 0.03125f  // DIM^-0.5

// ---------------------------------------------------------------------------
// 16x32 (A) / 32x16 (B) bf16 fragment loader.
// K-chunk scheme (ISA 7.12.2): lane&15 selects row (A: M) / col (B: N);
// elems 0..7  = K[khalf + 0..7],  elems 8..15 = K[16 + khalf + 0..7],
// khalf = 8 * (lane >= 16).  Two 16-byte loads per fragment.
// ---------------------------------------------------------------------------
__device__ __forceinline__ v16bf load_frag(const bf16* p, int ld, int lane) {
  const bf16* r = p + (lane & 15) * ld + ((lane >> 4) << 3);
  v8bf lo = *(const v8bf*)(r);
  v8bf hi = *(const v8bf*)(r + 16);
  v16bf f;
#pragma unroll
  for (int i = 0; i < 8; ++i) { f[i] = lo[i]; f[i + 8] = hi[i]; }
  return f;
}

__device__ __forceinline__ v8f wmma_bf16(v16bf a, v16bf b, v8f c) {
  // D(f32 16x16) = A(bf16 16x32) * B(bf16 32x16) + C
  return __builtin_amdgcn_wmma_f32_16x16x32_bf16(false, a, false, b,
                                                 (short)0, c, false, false);
}

// 32(M) x 64(N) wave tile over K, A row-major [M][K], Bt row-major [N][K].
__device__ __forceinline__ void gemm32x64(const bf16* __restrict__ A, int lda,
                                          const bf16* __restrict__ Bt, int ldb,
                                          int m0, int n0, int K, int lane,
                                          v8f acc[2][4]) {
  for (int k0 = 0; k0 < K; k0 += 32) {
    v16bf a0 = load_frag(A + (size_t)m0 * lda + k0, lda, lane);
    v16bf a1 = load_frag(A + (size_t)(m0 + 16) * lda + k0, lda, lane);
#pragma unroll
    for (int j = 0; j < 4; ++j) {
      v16bf b = load_frag(Bt + (size_t)(n0 + 16 * j) * ldb + k0, ldb, lane);
      acc[0][j] = wmma_bf16(a0, b, acc[0][j]);
      acc[1][j] = wmma_bf16(a1, b, acc[1][j]);
    }
  }
}

// ---------------------------------------------------------------------------
// Prep kernels
// ---------------------------------------------------------------------------
__global__ __launch_bounds__(256) void k_cvt_bf16(const float* __restrict__ s,
                                                  bf16* __restrict__ d, int n) {
  int i = blockIdx.x * 256 + threadIdx.x;
  if (i < n) d[i] = (bf16)s[i];
}

// wT[n*R + k] = w[k*C + n]  (w is [R=K][C=N] row-major)
__global__ __launch_bounds__(256) void k_transpose_bf16(const float* __restrict__ w,
                                                        bf16* __restrict__ wT,
                                                        int R, int C) {
  int idx = blockIdx.x * 256 + threadIdx.x;
  if (idx < R * C) {
    int n = idx / R;
    int k = idx - n * R;
    wT[idx] = (bf16)w[(size_t)k * C + n];
  }
}

// ---------------------------------------------------------------------------
// QKV projection: [8192 x 1024] x [1024 x 3072] -> scatter Q(scaled)/K/V^T bf16
// ---------------------------------------------------------------------------
__global__ __launch_bounds__(256) void k_qkv(const bf16* __restrict__ xb,
                                             const bf16* __restrict__ wT,
                                             bf16* __restrict__ Qb,
                                             bf16* __restrict__ Kb,
                                             bf16* __restrict__ VTb) {
  const int lane = threadIdx.x & 31;
  const int wid = blockIdx.x * 8 + (threadIdx.x >> 5);  // 12288 waves
  const int m0 = (wid & 255) * 32;
  const int n0 = (wid >> 8) * 64;
  v8f z = {};
  v8f acc[2][4];
#pragma unroll
  for (int a = 0; a < 2; ++a)
#pragma unroll
    for (int j = 0; j < 4; ++j) acc[a][j] = z;

  gemm32x64(xb, DIM, wT, DIM, m0, n0, DIM, lane, acc);

  const int colL = lane & 15;
  const int rb = (lane >> 4) << 3;  // C/D layout: rows 0..7 low half, 8..15 high
#pragma unroll
  for (int a = 0; a < 2; ++a)
#pragma unroll
    for (int j = 0; j < 4; ++j)
#pragma unroll
      for (int i = 0; i < 8; ++i) {
        int row = m0 + a * 16 + rb + i;
        int col = n0 + j * 16 + colL;
        float v = acc[a][j][i];
        int part = col >> 10, hd = col & 1023, h = hd >> 6, d = hd & 63;
        int bb = row >> 11, n = row & 2047;
        int bh = bb * HEADS + h;
        if (part == 0)
          Qb[((size_t)bh * NSEQ + n) * HD + d] = (bf16)(v * SCALE);
        else if (part == 1)
          Kb[((size_t)bh * NSEQ + n) * HD + d] = (bf16)v;
        else
          VTb[((size_t)bh * HD + d) * NSEQ + n] = (bf16)v;
      }
}

// ---------------------------------------------------------------------------
// Flash attention: one wave = 16-row strip of one (b,h). Online softmax fp32.
// ---------------------------------------------------------------------------
__global__ __launch_bounds__(256) void k_attn(const bf16* __restrict__ Qb,
                                              const bf16* __restrict__ Kb,
                                              const bf16* __restrict__ VTb,
                                              bf16* __restrict__ attnb) {
  __shared__ __align__(16) bf16 psh[8][16][32];  // per-wave P tile (16x32)
  const int lane = threadIdx.x & 31;
  const int w = threadIdx.x >> 5;
  const int wid = blockIdx.x * 8 + w;  // 8192 waves total
  const int bh = wid >> 7;             // 0..63
  const int i0 = (wid & 127) << 4;     // query-row strip
  const int b = bh >> 4, h = bh & 15;

  const bf16* Q = Qb + (size_t)bh * NSEQ * HD;
  const bf16* Kp = Kb + (size_t)bh * NSEQ * HD;
  const bf16* VT = VTb + (size_t)bh * HD * NSEQ;

  const v16bf q0 = load_frag(Q + (size_t)i0 * HD + 0, HD, lane);
  const v16bf q1 = load_frag(Q + (size_t)i0 * HD + 32, HD, lane);

  const int colL = lane & 15;
  const int rb = (lane >> 4) << 3;

  v8f z = {};
  float m[8], l[8];
  v8f o[4];
#pragma unroll
  for (int i = 0; i < 8; ++i) { m[i] = -1e30f; l[i] = 0.f; }
#pragma unroll
  for (int nt = 0; nt < 4; ++nt) o[nt] = z;

  for (int jb = 0; jb < NSEQ; jb += 32) {
    // S tile 16x32: two 16x16 WMMA chains over d (K=64 = 2 x 32)
    v16bf k00 = load_frag(Kp + (size_t)jb * HD + 0, HD, lane);
    v16bf k01 = load_frag(Kp + (size_t)jb * HD + 32, HD, lane);
    v16bf k10 = load_frag(Kp + (size_t)(jb + 16) * HD + 0, HD, lane);
    v16bf k11 = load_frag(Kp + (size_t)(jb + 16) * HD + 32, HD, lane);
    v8f s0 = z, s1 = z;
    s0 = wmma_bf16(q0, k00, s0);
    s0 = wmma_bf16(q1, k01, s0);
    s1 = wmma_bf16(q0, k10, s1);
    s1 = wmma_bf16(q1, k11, s1);

    // Online softmax: row r = rb+i lives in acc elem i across 16 lanes of a half.
#pragma unroll
    for (int i = 0; i < 8; ++i) {
      float cand = fmaxf(s0[i], s1[i]);
#pragma unroll
      for (int off = 1; off < 16; off <<= 1)
        cand = fmaxf(cand, __shfl_xor(cand, off, 32));
      float mn = fmaxf(m[i], cand);
      float corr = __expf(m[i] - mn);
      m[i] = mn;
      float p0 = __expf(s0[i] - mn);
      float p1 = __expf(s1[i] - mn);
      float ls = p0 + p1;
#pragma unroll
      for (int off = 1; off < 16; off <<= 1)
        ls += __shfl_xor(ls, off, 32);
      l[i] = l[i] * corr + ls;
#pragma unroll
      for (int nt = 0; nt < 4; ++nt) o[nt][i] *= corr;
      psh[w][rb + i][colL] = (bf16)p0;
      psh[w][rb + i][colL + 16] = (bf16)p1;
    }
    // intra-wave LDS RAW: DS ops are in-order per wave; fence the compiler + cnt
    asm volatile("s_wait_dscnt 0" ::: "memory");

    // Re-layout P (C/D layout -> A fragment) via LDS, then O += P @ V
    v16bf pa = load_frag(&psh[w][0][0], 32, lane);
#pragma unroll
    for (int nt = 0; nt < 4; ++nt) {
      v16bf vf = load_frag(VT + (size_t)(nt * 16) * NSEQ + jb, NSEQ, lane);
      o[nt] = wmma_bf16(pa, vf, o[nt]);
    }
  }

  float rl[8];
#pragma unroll
  for (int i = 0; i < 8; ++i) rl[i] = 1.f / l[i];
#pragma unroll
  for (int nt = 0; nt < 4; ++nt)
#pragma unroll
    for (int i = 0; i < 8; ++i) {
      int row = i0 + rb + i;
      int col = h * HD + nt * 16 + colL;
      attnb[((size_t)b * NSEQ + row) * DIM + col] = (bf16)(o[nt][i] * rl[i]);
    }
}

// ---------------------------------------------------------------------------
// Output projection: [8192 x 1024] x [1024 x 1024] + bias -> fp32 out
// ---------------------------------------------------------------------------
__global__ __launch_bounds__(256) void k_proj(const bf16* __restrict__ attnb,
                                              const bf16* __restrict__ wT,
                                              const float* __restrict__ bias,
                                              float* __restrict__ out) {
  const int lane = threadIdx.x & 31;
  const int wid = blockIdx.x * 8 + (threadIdx.x >> 5);  // 4096 waves
  const int m0 = (wid & 255) * 32;
  const int n0 = (wid >> 8) * 64;
  v8f z = {};
  v8f acc[2][4];
#pragma unroll
  for (int a = 0; a < 2; ++a)
#pragma unroll
    for (int j = 0; j < 4; ++j) acc[a][j] = z;

  gemm32x64(attnb, DIM, wT, DIM, m0, n0, DIM, lane, acc);

  const int colL = lane & 15;
  const int rb = (lane >> 4) << 3;
#pragma unroll
  for (int a = 0; a < 2; ++a)
#pragma unroll
    for (int j = 0; j < 4; ++j)
#pragma unroll
      for (int i = 0; i < 8; ++i) {
        int row = m0 + a * 16 + rb + i;
        int col = n0 + j * 16 + colL;
        out[(size_t)row * DIM + col] = acc[a][j][i] + bias[col];
      }
}

// ---------------------------------------------------------------------------
extern "C" void kernel_launch(void* const* d_in, const int* in_sizes, int n_in,
                              void* d_out, int out_size, void* d_ws, size_t ws_size,
                              hipStream_t stream) {
  const float* x = (const float*)d_in[0];      // [4,2048,1024]
  const float* w_qkv = (const float*)d_in[1];  // [1024,3072]
  const float* w_out = (const float*)d_in[2];  // [1024,1024]
  const float* b_out = (const float*)d_in[3];  // [1024]
  float* out = (float*)d_out;

  char* ws = (char*)d_ws;
  size_t off = 0;
  auto carve = [&](size_t bytes) {
    void* p = ws + off;
    off += (bytes + 255) & ~(size_t)255;
    return p;
  };
  const size_t nX = (size_t)NB * NSEQ * DIM;           // 8,388,608
  const size_t nH = (size_t)NB * HEADS * NSEQ * HD;    // 8,388,608
  bf16* xb    = (bf16*)carve(nX * 2);                  // 16 MB
  bf16* wqkvT = (bf16*)carve((size_t)3 * DIM * DIM * 2);
  bf16* woutT = (bf16*)carve((size_t)DIM * DIM * 2);
  bf16* Qb    = (bf16*)carve(nH * 2);
  bf16* Kb    = (bf16*)carve(nH * 2);
  bf16* VTb   = (bf16*)carve(nH * 2);
  bf16* attnb = (bf16*)carve(nX * 2);                  // total ~88 MB

  k_cvt_bf16<<<(int)((nX + 255) / 256), 256, 0, stream>>>(x, xb, (int)nX);
  k_transpose_bf16<<<(3 * DIM * DIM + 255) / 256, 256, 0, stream>>>(w_qkv, wqkvT, DIM, 3 * DIM);
  k_transpose_bf16<<<(DIM * DIM + 255) / 256, 256, 0, stream>>>(w_out, woutT, DIM, DIM);

  k_qkv<<<1536, 256, 0, stream>>>(xb, wqkvT, Qb, Kb, VTb);   // 12288 waves
  k_attn<<<1024, 256, 0, stream>>>(Qb, Kb, VTb, attnb);      // 8192 waves
  k_proj<<<512, 256, 0, stream>>>(attnb, woutT, b_out, out); // 4096 waves
}
